// BiLevelRoutingAttention_68049461837846
// MI455X (gfx1250) — matrix operand order repaired and checked by reading, not compiled, not run on hardware
//
#include <hip/hip_runtime.h>

typedef __attribute__((ext_vector_type(16))) _Float16 v16h;
typedef __attribute__((ext_vector_type(8)))  float    v8f;
typedef __attribute__((ext_vector_type(2)))  _Float16 h2v;
typedef __attribute__((ext_vector_type(2)))  float    f2v;

#define DIMC   256
#define NHEADS 8
#define HD     32
#define WSZ    8
#define GH     8
#define GW     8
#define RR     64           // regions per head-batch
#define TT     64           // tokens per window
#define TOPKN  4
#define BATCH  8
#define HH     64
#define WWW    64
#define BHN    (BATCH*NHEADS)   // 64
#define NPOS   (BATCH*HH*WWW)   // 32768
#define KT     (TOPKN*TT)       // 256
#define SCALEF 0.17677669529663687f

// ---------------------------------------------------------------------------
// Fragment helpers (wave32; A/B use this code's contiguous k order)
//   A 16x32 f16 : lane row = lane&15 ; vgpr v holds K = (lane>=16?16:0) + 2v, +1
//   B 32x16 f16 : lane col = lane&15 ; vgpr v holds K = (lane>=16?16:0) + 2v, +1
//   C/D 16x16 f32: lane col = lane&15 ; vgpr i -> row i + 8*(lane>=16)
// ---------------------------------------------------------------------------
__device__ __forceinline__ int a_kidx(int v, int halfsel) {
    return halfsel * 16 + 2 * v;
}
__device__ __forceinline__ int b_kidx(int v, int halfsel) {
    return halfsel * 16 + 2 * v;
}

// ---------------------------------------------------------------------------
// Kernel 1: qkv 1x1 conv as GEMM (M=32768 positions, N=768, K=256).
// Each wave computes a 16x64 tile (4 accumulators share one A fragment per
// k-step -> 4x less x re-read). Window-partition scatter fused into epilogue.
// ---------------------------------------------------------------------------
__global__ __launch_bounds__(32)
void qkv_wmma(const float* __restrict__ x, const float* __restrict__ w_qkv,
              const float* __restrict__ b_qkv,
              _Float16* __restrict__ qw, _Float16* __restrict__ kw,
              _Float16* __restrict__ vw)
{
    const int lane    = threadIdx.x & 31;
    const int col     = lane & 15;
    const int halfsel = lane >> 4;
    const int m0      = blockIdx.x * 16;   // position tile
    const int n0      = blockIdx.y * 64;   // output-channel super-tile

    v8f acc[4];
    #pragma unroll
    for (int nt2 = 0; nt2 < 4; ++nt2) {
        float bias = b_qkv[n0 + nt2 * 16 + col];
        #pragma unroll
        for (int i = 0; i < 8; ++i) acc[nt2][i] = bias;
    }

    // A row for this lane: position arow -> x[b, :, h, w], channel stride 4096
    const int arow = m0 + (lane & 15);
    const int bb   = arow >> 12;            // /4096
    const int hw   = arow & 4095;
    const float* xrow = x + (size_t)bb * DIMC * 4096 + hw;
    const float* wrow[4];
    #pragma unroll
    for (int nt2 = 0; nt2 < 4; ++nt2)
        wrow[nt2] = w_qkv + (size_t)(n0 + nt2 * 16 + col) * DIMC;

    #pragma unroll
    for (int kk = 0; kk < DIMC; kk += 32) {
        if (kk + 32 < DIMC)
            __builtin_prefetch(xrow + (size_t)(kk + 32) * 4096, 0, 0);
        v16h a;
        #pragma unroll
        for (int v = 0; v < 8; ++v) {
            int ka = kk + a_kidx(v, halfsel);
            a[2 * v]     = (_Float16)xrow[(size_t)ka * 4096];
            a[2 * v + 1] = (_Float16)xrow[(size_t)(ka + 1) * 4096];
        }
        #pragma unroll
        for (int nt2 = 0; nt2 < 4; ++nt2) {
            v16h bf;
            #pragma unroll
            for (int v = 0; v < 8; ++v) {
                f2v wp = *(const f2v*)(wrow[nt2] + kk + b_kidx(v, halfsel));
                bf[2 * v]     = (_Float16)wp.x;
                bf[2 * v + 1] = (_Float16)wp.y;
            }
            acc[nt2] = __builtin_amdgcn_wmma_f32_16x16x32_f16(
                false, a, false, bf, (short)0, acc[nt2], false, false);
        }
    }

    // epilogue: scatter into windowed layout
    #pragma unroll
    for (int nt2 = 0; nt2 < 4; ++nt2) {
        const int o     = n0 + nt2 * 16 + col;
        const int which = o >> 8;               // 0:q 1:k 2:v
        const int head  = (o & 255) >> 5;
        const int d     = o & 31;
        _Float16* dst = (which == 0) ? qw : ((which == 1) ? kw : vw);
        #pragma unroll
        for (int i = 0; i < 8; ++i) {
            int m   = m0 + i + 8 * halfsel;
            int b2  = m >> 12;
            int hw2 = m & 4095;
            int h = hw2 >> 6, w = hw2 & 63;
            int r = (h >> 3) * GW + (w >> 3);
            int t = (h & 7) * WSZ + (w & 7);
            int bh = b2 * NHEADS + head;
            dst[(((size_t)bh * RR + r) * TT + t) * HD + d] = (_Float16)acc[nt2][i];
        }
    }
}

// ---------------------------------------------------------------------------
// Kernel 2: region-level mean pooling  q_r/k_r [BH, R, HD] f32
// ---------------------------------------------------------------------------
__global__ __launch_bounds__(32)
void region_means(const _Float16* __restrict__ qw, const _Float16* __restrict__ kw,
                  float* __restrict__ q_r, float* __restrict__ k_r)
{
    int br = blockIdx.x;          // bh*R + r
    int d  = threadIdx.x;
    const _Float16* qp = qw + (size_t)br * TT * HD + d;
    const _Float16* kp = kw + (size_t)br * TT * HD + d;
    float sq = 0.f, sk = 0.f;
    #pragma unroll 8
    for (int t = 0; t < TT; ++t) {
        sq += (float)qp[t * HD];
        sk += (float)kp[t * HD];
    }
    q_r[(size_t)br * HD + d] = sq * (1.0f / TT);
    k_r[(size_t)br * HD + d] = sk * (1.0f / TT);
}

// ---------------------------------------------------------------------------
// Kernel 3: routing logits + top-4 region indices (tiny: 4096 x 64 argmax)
// ---------------------------------------------------------------------------
__global__ __launch_bounds__(32)
void routing_topk(const float* __restrict__ q_r, const float* __restrict__ k_r,
                  int* __restrict__ topk)
{
    __shared__ float logits[RR];
    int bh   = blockIdx.x >> 6;
    int r    = blockIdx.x & 63;
    int lane = threadIdx.x;
    const float* q = q_r + ((size_t)bh * RR + r) * HD;
    for (int k = lane; k < RR; k += 32) {
        const float* kp = k_r + ((size_t)bh * RR + k) * HD;
        float s = 0.f;
        #pragma unroll
        for (int d = 0; d < HD; ++d) s += q[d] * kp[d];
        logits[k] = s * SCALEF;
    }
    __syncthreads();
    if (lane == 0) {
        int* out = topk + ((size_t)bh * RR + r) * TOPKN;
        #pragma unroll
        for (int j = 0; j < TOPKN; ++j) {
            float best = -1e30f; int bi = 0;
            for (int k = 0; k < RR; ++k)
                if (logits[k] > best) { best = logits[k]; bi = k; }
            out[j] = bi;
            logits[bi] = -1e30f;
        }
    }
}

// ---------------------------------------------------------------------------
// Kernel 4: token-level attention per (bh, r).
//   Q tile staged into LDS with global_load_async_to_lds_b128 (ASYNCcnt path)
//   S = Q(64x32) @ Ksel^T(32x256) * scale -> LDS
//   P = softmax(S)                        -> LDS (in place)
//   O = P(64x256) @ Vsel(256x32)          -> scattered to [B,C,H,W] f16
// 8 waves / block. LDS: 64x256 f32 = 64 KB (first 4 KB doubles as Q stage).
// ---------------------------------------------------------------------------
__global__ __launch_bounds__(256)
void attn_wmma(const _Float16* __restrict__ qw, const _Float16* __restrict__ kw,
               const _Float16* __restrict__ vw, const int* __restrict__ topk,
               _Float16* __restrict__ att)
{
    __shared__ float sc[TT][KT];          // 65536 bytes; sc is LDS offset 0

    const int bh = blockIdx.x >> 6;
    const int r  = blockIdx.x & 63;
    const int tid  = threadIdx.x;
    const int lane = tid & 31;
    const int wv   = tid >> 5;            // 0..7
    const int col     = lane & 15;
    const int halfsel = lane >> 4;

    const int* tki = topk + ((size_t)bh * RR + r) * TOPKN;
    const _Float16* qwin = qw + ((size_t)bh * RR + r) * TT * HD;

    // ---- async-stage Q (64x32 f16 = 4 KB) into LDS offset 0 ---------------
    {
        unsigned ldsoff = (unsigned)tid * 16u;         // sc lives at LDS offset 0
        const _Float16* gsrc = qwin + tid * 8;         // 16 bytes per thread
        asm volatile("global_load_async_to_lds_b128 %0, %1, off"
                     :: "v"(ldsoff), "v"(gsrc) : "memory");
        asm volatile("s_wait_asynccnt 0" ::: "memory");
    }
    __syncthreads();

    // ---- build per-wave Q fragment from LDS, then release the stage buffer
    const int mt = wv & 3;
    v16h aq;
    {
        const _Float16* qs = (const _Float16*)&sc[0][0];
        const int arow = mt * 16 + (lane & 15);
        #pragma unroll
        for (int v = 0; v < 8; ++v) {
            h2v qv = *(const h2v*)(qs + arow * HD + a_kidx(v, halfsel));
            aq[2 * v]     = qv.x;
            aq[2 * v + 1] = qv.y;
        }
    }
    __syncthreads();

    // ---- stage 1: S = Q @ Ksel^T * scale ----------------------------------
    // wave wv: M-tile mt = wv&3, N-tiles nt = (wv>>2)*8 + i, i=0..7
    #pragma unroll
    for (int i = 0; i < 8; ++i) {
        int nt = (wv >> 2) * 8 + i;
        int s  = nt * 16 + col;       // selected-token index 0..255
        int j  = s >> 6;              // which routed window
        int tt = s & 63;              // token within window
        const _Float16* kwin = kw + ((size_t)bh * RR + tki[j]) * TT * HD;
        v16h bf;
        #pragma unroll
        for (int v = 0; v < 8; ++v) {
            h2v kv = *(const h2v*)(kwin + tt * HD + b_kidx(v, halfsel));
            bf[2 * v]     = kv.x;
            bf[2 * v + 1] = kv.y;
        }
        v8f c = {};
        c = __builtin_amdgcn_wmma_f32_16x16x32_f16(false, aq, false, bf,
                                                   (short)0, c, false, false);
        #pragma unroll
        for (int e = 0; e < 8; ++e)
            sc[mt * 16 + e + 8 * halfsel][nt * 16 + col] = c[e] * SCALEF;
    }
    __syncthreads();

    // ---- stage 2: row softmax over 256 (4 adjacent lanes per row) ---------
    {
        int row = tid >> 2;
        int seg = tid & 3;
        float* prow = &sc[row][0];
        float mx = -1e30f;
        #pragma unroll 16
        for (int jx = seg * 64; jx < seg * 64 + 64; ++jx) mx = fmaxf(mx, prow[jx]);
        mx = fmaxf(mx, __shfl_xor(mx, 1, 32));
        mx = fmaxf(mx, __shfl_xor(mx, 2, 32));
        float sum = 0.f;
        #pragma unroll 16
        for (int jx = seg * 64; jx < seg * 64 + 64; ++jx) {
            float e = __expf(prow[jx] - mx);
            prow[jx] = e;
            sum += e;
        }
        sum += __shfl_xor(sum, 1, 32);
        sum += __shfl_xor(sum, 2, 32);
        float inv = 1.0f / sum;
        #pragma unroll 16
        for (int jx = seg * 64; jx < seg * 64 + 64; ++jx) prow[jx] *= inv;
    }
    __syncthreads();

    // ---- stage 3: O = P @ Vsel --------------------------------------------
    // 8 output tiles (mt 0..3) x (nt 0..1) -> one per wave. K = 256.
    {
        const int nt   = wv >> 2;
        const int arow = mt * 16 + (lane & 15);
        const int d    = nt * 16 + col;
        v8f c = {};
        #pragma unroll
        for (int kk = 0; kk < 8; ++kk) {
            v16h a, bf;
            #pragma unroll
            for (int v = 0; v < 8; ++v) {
                f2v pv = *(const f2v*)&sc[arow][kk * 32 + a_kidx(v, halfsel)];
                a[2 * v]     = (_Float16)pv.x;
                a[2 * v + 1] = (_Float16)pv.y;
            }
            // B rows are selected tokens s; all 16 k's of this lane-half sit
            // in one routed window (32-blocks never straddle a 64 boundary)
            int sbase = kk * 32 + halfsel * 16;
            int j = sbase >> 6;
            const _Float16* vwin = vw + ((size_t)bh * RR + tki[j]) * TT * HD;
            #pragma unroll
            for (int v = 0; v < 8; ++v) {
                int s0 = sbase + 2 * v;
                bf[2 * v]     = vwin[(s0 & 63) * HD + d];
                bf[2 * v + 1] = vwin[((s0 + 1) & 63) * HD + d];
            }
            c = __builtin_amdgcn_wmma_f32_16x16x32_f16(false, a, false, bf,
                                                       (short)0, c, false, false);
        }
        // scatter: window reverse into [B, C, H, W] f16
        const int head = bh & 7;
        const int b2   = bh >> 3;
        const int gh   = r >> 3, gw = r & 7;
        const int cch  = head * HD + d;
        #pragma unroll
        for (int e = 0; e < 8; ++e) {
            int t = mt * 16 + e + 8 * halfsel;
            int h = gh * WSZ + (t >> 3);
            int w = gw * WSZ + (t & 7);
            att[(((size_t)b2 * DIMC + cch) * HH + h) * WWW + w] = (_Float16)c[e];
        }
    }
}

// ---------------------------------------------------------------------------
// Kernel 5: output projection GEMM (M=32768, N=256, K=256), f32 output.
// 16x64 tile per wave (4 accumulators share A fragment).
// ---------------------------------------------------------------------------
__global__ __launch_bounds__(32)
void proj_wmma(const _Float16* __restrict__ att, const float* __restrict__ w_proj,
               const float* __restrict__ b_proj, float* __restrict__ out)
{
    const int lane    = threadIdx.x & 31;
    const int col     = lane & 15;
    const int halfsel = lane >> 4;
    const int m0      = blockIdx.x * 16;
    const int n0      = blockIdx.y * 64;

    v8f acc[4];
    #pragma unroll
    for (int nt2 = 0; nt2 < 4; ++nt2) {
        float bias = b_proj[n0 + nt2 * 16 + col];
        #pragma unroll
        for (int i = 0; i < 8; ++i) acc[nt2][i] = bias;
    }

    const int arow = m0 + (lane & 15);
    const int bb   = arow >> 12;
    const int hw   = arow & 4095;
    const _Float16* arowp = att + (size_t)bb * DIMC * 4096 + hw;
    const float* wrow[4];
    #pragma unroll
    for (int nt2 = 0; nt2 < 4; ++nt2)
        wrow[nt2] = w_proj + (size_t)(n0 + nt2 * 16 + col) * DIMC;

    #pragma unroll
    for (int kk = 0; kk < DIMC; kk += 32) {
        if (kk + 32 < DIMC)
            __builtin_prefetch(arowp + (size_t)(kk + 32) * 4096, 0, 0);
        v16h a;
        #pragma unroll
        for (int v = 0; v < 8; ++v) {
            int ka = kk + a_kidx(v, halfsel);
            a[2 * v]     = arowp[(size_t)ka * 4096];
            a[2 * v + 1] = arowp[(size_t)(ka + 1) * 4096];
        }
        #pragma unroll
        for (int nt2 = 0; nt2 < 4; ++nt2) {
            v16h bf;
            #pragma unroll
            for (int v = 0; v < 8; ++v) {
                f2v wp = *(const f2v*)(wrow[nt2] + kk + b_kidx(v, halfsel));
                bf[2 * v]     = (_Float16)wp.x;
                bf[2 * v + 1] = (_Float16)wp.y;
            }
            acc[nt2] = __builtin_amdgcn_wmma_f32_16x16x32_f16(
                false, a, false, bf, (short)0, acc[nt2], false, false);
        }
    }

    #pragma unroll
    for (int nt2 = 0; nt2 < 4; ++nt2) {
        #pragma unroll
        for (int i = 0; i < 8; ++i) {
            int m   = m0 + i + 8 * halfsel;
            int b2  = m >> 12;
            int hw2 = m & 4095;
            out[((size_t)b2 * DIMC + (n0 + nt2 * 16 + col)) * 4096 + hw2] =
                acc[nt2][i];
        }
    }
}

// ---------------------------------------------------------------------------
// Workspace layout (bytes):
//   qw  f16 [64,64,64,32]  @ 0          (16 MiB)
//   kw  f16                @ 16777216   (16 MiB)
//   vw  f16                @ 33554432   (16 MiB)
//   q_r f32 [64,64,32]     @ 50331648   (512 KiB)
//   k_r f32                @ 50855936   (512 KiB)
//   topk i32 [64,64,4]     @ 51380224   (64 KiB)
//   att f16 [8,256,64,64]  @ 51445760   (16 MiB)   total ~65.1 MiB
// ---------------------------------------------------------------------------
extern "C" void kernel_launch(void* const* d_in, const int* in_sizes, int n_in,
                              void* d_out, int out_size, void* d_ws, size_t ws_size,
                              hipStream_t stream)
{
    const float* x      = (const float*)d_in[0];
    const float* w_qkv  = (const float*)d_in[1];
    const float* b_qkv  = (const float*)d_in[2];
    const float* w_proj = (const float*)d_in[3];
    const float* b_proj = (const float*)d_in[4];
    float* out = (float*)d_out;

    char* ws = (char*)d_ws;
    _Float16* qw  = (_Float16*)(ws + 0);
    _Float16* kw  = (_Float16*)(ws + 16777216u);
    _Float16* vw  = (_Float16*)(ws + 33554432u);
    float*    q_r = (float*)(ws + 50331648u);
    float*    k_r = (float*)(ws + 50855936u);
    int*      tpk = (int*)(ws + 51380224u);
    _Float16* att = (_Float16*)(ws + 51445760u);

    qkv_wmma<<<dim3(NPOS / 16, (3 * DIMC) / 64), 32, 0, stream>>>(
        x, w_qkv, b_qkv, qw, kw, vw);
    region_means<<<BHN * RR, 32, 0, stream>>>(qw, kw, q_r, k_r);
    routing_topk<<<BHN * RR, 32, 0, stream>>>(q_r, k_r, tpk);
    attn_wmma<<<BHN * RR, 256, 0, stream>>>(qw, kw, vw, tpk, att);
    proj_wmma<<<dim3(NPOS / 16, DIMC / 64), 32, 0, stream>>>(
        att, w_proj, b_proj, out);
}
